// MultiHeadAttention_77309412073
// MI455X (gfx1250) — compile-verified, matrix operand order checked by reading, not run on hardware
//
#include <hip/hip_runtime.h>
#include <hip/hip_bf16.h>

// ---------------------------------------------------------------------------
// MI455X (gfx1250) MHA forward.
//  - bf16 WMMA (v_wmma_f32_16x16x32_bf16) for all GEMMs, fp32 accumulate
//  - fused scores -> softmax -> attn@V kernel: each workgroup owns a
//    128x2048 strip of one (b,h); raw scores written once, normalized in
//    L2, and consumed immediately => ~1x HBM traffic on the 536MB attn
//    tensor instead of 4x.
//  - V-head tiles staged into LDS via the Tensor Data Mover
//    (tensor_load_to_lds + s_wait_tensorcnt), double buffered.
// ---------------------------------------------------------------------------

typedef __attribute__((ext_vector_type(16))) __bf16 v16bf;
typedef __attribute__((ext_vector_type(8)))  __bf16 bf16x8;
typedef __attribute__((ext_vector_type(4)))  __bf16 bf16x4;
typedef __attribute__((ext_vector_type(8)))  float  v8f;
typedef __attribute__((ext_vector_type(4)))  unsigned int u32x4;
typedef __attribute__((ext_vector_type(8)))  int i32x8;
typedef __attribute__((ext_vector_type(4)))  int i32x4;

#if defined(__gfx1250__) && __has_builtin(__builtin_amdgcn_tensor_load_to_lds) && \
    __has_builtin(__builtin_amdgcn_s_wait_tensorcnt)
#define ATTN_TDM 1
#else
#define ATTN_TDM 0
#endif

union ABFrag { v16bf v; bf16x8 h[2]; };

__device__ inline bf16x8 ld8frag(const __bf16* p) {
    return *(const bf16x8*)p;                       // 16B load
}
__device__ inline bf16x8 ld8frag(const float* p) {  // fp32 -> bf16 on load
    const float4 f0 = *(const float4*)p;
    const float4 f1 = *(const float4*)(p + 4);
    bf16x8 r;
    r[0] = (__bf16)f0.x; r[1] = (__bf16)f0.y; r[2] = (__bf16)f0.z; r[3] = (__bf16)f0.w;
    r[4] = (__bf16)f1.x; r[5] = (__bf16)f1.y; r[6] = (__bf16)f1.z; r[7] = (__bf16)f1.w;
    return r;
}

// ---------------------------------------------------------------------------
// Generic batched NT GEMM (projections / output projection).
// Block 256 = 8 waves, tile 128(M) x 64(N), wave tile 32x32, K-step 32.
// ---------------------------------------------------------------------------
template <typename TA, typename TC>
__global__ __launch_bounds__(256) void gemm_nt_wmma(
    const TA* __restrict__ A, const __bf16* __restrict__ Bm, TC* __restrict__ C,
    int M, int N, int K, int lda, int ldb, int ldc,
    float scale, const float* __restrict__ bias)
{
    const int lane  = threadIdx.x & 31;
    const int wave  = threadIdx.x >> 5;
    const int wM    = wave & 3;
    const int wN    = wave >> 2;
    const int mBase = blockIdx.y * 128 + wM * 32;
    const int nBase = blockIdx.x * 64  + wN * 32;
    const int r     = lane & 15;
    const int sel   = lane >> 4;

    v8f acc[2][2] = {};

    for (int kk = 0; kk < K; kk += 32) {
        ABFrag a[2], bb[2];
#pragma unroll
        for (int t = 0; t < 2; ++t) {
            const TA* ap = A + (size_t)(mBase + t * 16 + r) * lda + kk + sel * 8;
            a[t].h[0] = ld8frag(ap);
            a[t].h[1] = ld8frag(ap + 16);
            const __bf16* bp = Bm + (size_t)(nBase + t * 16 + r) * ldb + kk + sel * 8;
            bb[t].h[0] = ld8frag(bp);
            bb[t].h[1] = ld8frag(bp + 16);
        }
#pragma unroll
        for (int mt = 0; mt < 2; ++mt)
#pragma unroll
            for (int nt = 0; nt < 2; ++nt)
                acc[mt][nt] = __builtin_amdgcn_wmma_f32_16x16x32_bf16(
                    false, a[mt].v, false, bb[nt].v,
                    (short)0, acc[mt][nt], false, false);
    }

#pragma unroll
    for (int mt = 0; mt < 2; ++mt) {
#pragma unroll
        for (int nt = 0; nt < 2; ++nt) {
            const int col = nBase + nt * 16 + r;
            const float bv = bias ? bias[col] : 0.0f;
#pragma unroll
            for (int q = 0; q < 8; ++q) {
                const int row = mBase + mt * 16 + q + sel * 8;
                C[(size_t)row * ldc + col] = (TC)(acc[mt][nt][q] * scale + bv);
            }
        }
    }
}

// ---------------------------------------------------------------------------
// V-head chunk loader: 64 rows (n) x 64 cols (k) bf16 = 8 KB into LDS.
// TDM path: one tensor_load_to_lds per workgroup (wave 0), TENSORcnt sync.
// Fallback: cooperative synchronous copy (ds_store path).
// ---------------------------------------------------------------------------
__device__ inline void v_chunk_issue(const __bf16* gsrc, __bf16* dst,
                                     int tid, int wave)
{
#if ATTN_TDM
    if (wave == 0) {
        u32x4 g0;
        g0[0] = 1u;                                        // count=1, user mode
        g0[1] = (unsigned)(size_t)(void*)dst;              // LDS byte address
        const unsigned long long ga = (unsigned long long)(size_t)gsrc;
        g0[2] = (unsigned)(ga & 0xFFFFFFFFu);
        g0[3] = (unsigned)((ga >> 32) & 0x01FFFFFFu) | 0x80000000u; // type=2
        i32x8 g1;
        g1[0] = 0x00010000;            // workgroup_mask=0, data_size=1 (2B)
        g1[1] = (int)(2048u << 16);    // tensor_dim0 = S (lo16 at [31:16])
        g1[2] = (int)(64u << 16);      // tensor_dim0 hi=0 | tensor_dim1 = 64
        g1[3] = (int)(64u << 16);      // tensor_dim1 hi=0 | tile_dim0 = 64
        g1[4] = 64;                    // tile_dim1 = 64, tile_dim2 = 0
        g1[5] = 2048;                  // tensor_dim0_stride = S (lo32)
        g1[6] = 0;
        g1[7] = 0;
        const i32x4 z4 = {0, 0, 0, 0};
#if defined(__clang_major__) && (__clang_major__ >= 23)
        const i32x8 z8 = {0, 0, 0, 0, 0, 0, 0, 0};
        __builtin_amdgcn_tensor_load_to_lds(g0, g1, z4, z4, z8, 0);
#else
        __builtin_amdgcn_tensor_load_to_lds(g0, g1, z4, z4, 0);
#endif
    }
#else
    const int n  = tid >> 2;             // 64 rows
    const int ko = (tid & 3) * 16;       // 4 x 16 elems = 64 k
    const __bf16* s = gsrc + (size_t)n * 2048 + ko;
    const bf16x8 a = *(const bf16x8*)s;
    const bf16x8 b = *(const bf16x8*)(s + 8);
    *(bf16x8*)(dst + n * 64 + ko)     = a;
    *(bf16x8*)(dst + n * 64 + ko + 8) = b;
#endif
}

__device__ inline void v_chunk_wait(int wave)
{
#if ATTN_TDM
    if (wave == 0) __builtin_amdgcn_s_wait_tensorcnt(0);
#endif
    (void)wave;
}

// ---------------------------------------------------------------------------
// Fused attention: scores (WMMA) -> softmax (shuffle) -> ctx (WMMA, V in LDS)
// grid = (1, S/128, B*H); block = 256 (8 waves).
// ---------------------------------------------------------------------------
__global__ __launch_bounds__(256) void attn_fused(
    const __bf16* __restrict__ Qp, const __bf16* __restrict__ Kp,
    const __bf16* __restrict__ Vt, float* __restrict__ attn,
    __bf16* __restrict__ ctx, const int* __restrict__ mask, float scale)
{
    constexpr int S = 2048, D = 1024, H = 16, DK = 64;
    const int z = blockIdx.z, b = z / H, h = z % H;
    const __bf16* Qz = Qp + (size_t)b * S * D + (size_t)h * DK;
    const __bf16* Kz = Kp + (size_t)b * S * D + (size_t)h * DK;
    const __bf16* Vz = Vt + (size_t)z * DK * S;
    float*        Az = attn + (size_t)z * S * S;
    __bf16*       Cz = ctx + (size_t)b * S * D + (size_t)h * DK;
    const int*    mz = mask + (size_t)b * S;

    const int tid  = threadIdx.x;
    const int lane = tid & 31, wave = tid >> 5;
    const int wM = wave & 3, wN = wave >> 2;
    const int r = lane & 15, sel = lane >> 4;
    const int rowBase = blockIdx.y * 128;

    __shared__ alignas(16) __bf16 vbuf[2][DK * 64];

    // ---------------- Phase 1: raw scaled+masked scores ----------------
    ABFrag aQ[2][2];                       // [m-tile][k-step], reused all n
#pragma unroll
    for (int mt = 0; mt < 2; ++mt)
#pragma unroll
        for (int ks = 0; ks < 2; ++ks) {
            const __bf16* p = Qz + (size_t)(rowBase + wM * 32 + mt * 16 + r) * D
                              + ks * 32 + sel * 8;
            aQ[mt][ks].h[0] = ld8frag(p);
            aQ[mt][ks].h[1] = ld8frag(p + 16);
        }

    for (int nb = 0; nb < S; nb += 64) {
        const int nW = nb + wN * 32;
        ABFrag bK[2][2];
#pragma unroll
        for (int nt = 0; nt < 2; ++nt)
#pragma unroll
            for (int ks = 0; ks < 2; ++ks) {
                const __bf16* p = Kz + (size_t)(nW + nt * 16 + r) * D
                                  + ks * 32 + sel * 8;
                bK[nt][ks].h[0] = ld8frag(p);
                bK[nt][ks].h[1] = ld8frag(p + 16);
            }
        v8f acc[2][2] = {};
#pragma unroll
        for (int ks = 0; ks < 2; ++ks)
#pragma unroll
            for (int mt = 0; mt < 2; ++mt)
#pragma unroll
                for (int nt = 0; nt < 2; ++nt)
                    acc[mt][nt] = __builtin_amdgcn_wmma_f32_16x16x32_bf16(
                        false, aQ[mt][ks].v, false, bK[nt][ks].v,
                        (short)0, acc[mt][nt], false, false);
#pragma unroll
        for (int mt = 0; mt < 2; ++mt)
#pragma unroll
            for (int nt = 0; nt < 2; ++nt) {
                const int col = nW + nt * 16 + r;
                const bool m0 = (mz[col] == 0);
#pragma unroll
                for (int q = 0; q < 8; ++q) {
                    const int row = rowBase + wM * 32 + mt * 16 + q + sel * 8;
                    float vv = acc[mt][nt][q] * scale;
                    if (m0) vv = -1.0e9f;
                    Az[(size_t)row * S + col] = vv;
                }
            }
    }
    __threadfence();
    __syncthreads();

    // ---------------- Phase 2: softmax, wave-per-row ----------------
    for (int rr = 0; rr < 16; ++rr) {
        const int row = rowBase + wave * 16 + rr;
        float4* p4 = (float4*)(Az + (size_t)row * S);
        float4 x[16];
        float lm = -3.4e38f;
#pragma unroll
        for (int i = 0; i < 16; ++i) {
            x[i] = p4[i * 32 + lane];
            lm = fmaxf(lm, fmaxf(fmaxf(x[i].x, x[i].y), fmaxf(x[i].z, x[i].w)));
        }
#pragma unroll
        for (int off = 16; off > 0; off >>= 1)
            lm = fmaxf(lm, __shfl_xor(lm, off, 32));
        float ls = 0.0f;
#pragma unroll
        for (int i = 0; i < 16; ++i) {
            x[i].x = __expf(x[i].x - lm); x[i].y = __expf(x[i].y - lm);
            x[i].z = __expf(x[i].z - lm); x[i].w = __expf(x[i].w - lm);
            ls += (x[i].x + x[i].y) + (x[i].z + x[i].w);
        }
#pragma unroll
        for (int off = 16; off > 0; off >>= 1)
            ls += __shfl_xor(ls, off, 32);
        const float inv = 1.0f / ls;
#pragma unroll
        for (int i = 0; i < 16; ++i) {
            x[i].x *= inv; x[i].y *= inv; x[i].z *= inv; x[i].w *= inv;
            p4[i * 32 + lane] = x[i];
        }
    }
    __threadfence();
    __syncthreads();

    // ---------------- Phase 3: ctx = probs @ V (V in LDS via TDM) ----------
    v8f acc[2][2] = {};
    const int mB2 = rowBase + wM * 32;
    const int nB2 = wN * 32;

    v_chunk_issue(Vz, &vbuf[0][0], tid, wave);
    v_chunk_wait(wave);
    __syncthreads();

    for (int c = 0; c < 32; ++c) {                    // 32 chunks of K=64
        const int cur = c & 1;
        if (c + 1 < 32)
            v_chunk_issue(Vz + (c + 1) * 64, &vbuf[1 - cur][0], tid, wave);
        const __bf16* vb = &vbuf[cur][0];
#pragma unroll
        for (int ks = 0; ks < 2; ++ks) {
            const int kg = c * 64 + ks * 32;
            ABFrag aP[2], bV[2];
#pragma unroll
            for (int t = 0; t < 2; ++t) {
                const float* ap = Az + (size_t)(mB2 + t * 16 + r) * S + kg + sel * 8;
                aP[t].h[0] = ld8frag(ap);
                aP[t].h[1] = ld8frag(ap + 16);
                const __bf16* bp = vb + (nB2 + t * 16 + r) * 64 + ks * 32 + sel * 8;
                bV[t].h[0] = *(const bf16x8*)bp;
                bV[t].h[1] = *(const bf16x8*)(bp + 16);
            }
#pragma unroll
            for (int mt = 0; mt < 2; ++mt)
#pragma unroll
                for (int nt = 0; nt < 2; ++nt)
                    acc[mt][nt] = __builtin_amdgcn_wmma_f32_16x16x32_bf16(
                        false, aP[mt].v, false, bV[nt].v,
                        (short)0, acc[mt][nt], false, false);
        }
        if (c + 1 < 32) v_chunk_wait(wave);
        __syncthreads();
    }

#pragma unroll
    for (int mt = 0; mt < 2; ++mt)
#pragma unroll
        for (int nt = 0; nt < 2; ++nt) {
            const int col = nB2 + nt * 16 + r;
#pragma unroll
            for (int q = 0; q < 8; ++q) {
                const int row = mB2 + mt * 16 + q + sel * 8;
                Cz[(size_t)row * D + col] = (__bf16)acc[mt][nt][q];
            }
        }
}

// ---------------------------------------------------------------------------
__global__ __launch_bounds__(256) void cvt_f32_bf16(
    const float* __restrict__ in, __bf16* __restrict__ out, size_t n4)
{
    const size_t i = (size_t)blockIdx.x * blockDim.x + threadIdx.x;
    if (i >= n4) return;
    const float4 f = ((const float4*)in)[i];
    bf16x4 o;
    o[0] = (__bf16)f.x; o[1] = (__bf16)f.y; o[2] = (__bf16)f.z; o[3] = (__bf16)f.w;
    ((bf16x4*)out)[i] = o;
}

__global__ __launch_bounds__(256) void transpose_v_heads(
    const __bf16* __restrict__ Vp, __bf16* __restrict__ Vt,
    int Bc, int Hc, int Sc, int Dc, int DKc)
{
    const size_t t = (size_t)blockIdx.x * blockDim.x + threadIdx.x;
    const size_t total = (size_t)Bc * Hc * DKc * Sc;
    if (t >= total) return;
    const int k = (int)(t % Sc);
    size_t rest = t / Sc;
    const int n = (int)(rest % DKc); rest /= DKc;
    const int h = (int)(rest % Hc);
    const int b = (int)(rest / Hc);
    Vt[t] = Vp[((size_t)b * Sc + k) * Dc + (size_t)h * DKc + n];
}

// ---------------------------------------------------------------------------
extern "C" void kernel_launch(void* const* d_in, const int* in_sizes, int n_in,
                              void* d_out, int out_size, void* d_ws, size_t ws_size,
                              hipStream_t stream)
{
    constexpr int Bc = 2, Sc = 2048, Dc = 1024, Hc = 16, DKc = 64;
    const float* q    = (const float*)d_in[0];
    const float* k    = (const float*)d_in[1];
    const float* v    = (const float*)d_in[2];
    const int*   mask = (const int*)  d_in[3];
    const float* wq   = (const float*)d_in[4];
    const float* wk   = (const float*)d_in[5];
    const float* wv   = (const float*)d_in[6];
    const float* wo   = (const float*)d_in[7];
    const float* bo   = (const float*)d_in[8];

    const size_t NE = (size_t)Bc * Sc * Dc;
    const size_t WE = (size_t)Dc * Dc;

    __bf16* wsb = (__bf16*)d_ws;
    __bf16* qbf = wsb;            __bf16* kbf = qbf + NE;  __bf16* vbf = kbf + NE;
    __bf16* wqb = vbf + NE;       __bf16* wkb = wqb + WE;
    __bf16* wvb = wkb + WE;       __bf16* wob = wvb + WE;
    __bf16* Qp  = wob + WE;       __bf16* Kp  = Qp + NE;   __bf16* Vp = Kp + NE;
    __bf16* Vt  = Vp + NE;        __bf16* ctx = Vt + NE;

    float* out  = (float*)d_out;
    float* attn = out + NE;

    const dim3 blk(256);

    auto cvt = [&](const float* src, __bf16* dst, size_t n) {
        const size_t n4 = n / 4;
        cvt_f32_bf16<<<dim3((unsigned)((n4 + 255) / 256)), blk, 0, stream>>>(src, dst, n4);
    };
    cvt(q, qbf, NE);  cvt(k, kbf, NE);  cvt(v, vbf, NE);
    cvt(wq, wqb, WE); cvt(wk, wkb, WE); cvt(wv, wvb, WE); cvt(wo, wob, WE);

    const dim3 gproj(Dc / 64, (Bc * Sc) / 128, 1);
    gemm_nt_wmma<__bf16, __bf16><<<gproj, blk, 0, stream>>>(
        qbf, wqb, Qp, Bc * Sc, Dc, Dc, Dc, Dc, Dc, 1.0f, nullptr);
    gemm_nt_wmma<__bf16, __bf16><<<gproj, blk, 0, stream>>>(
        kbf, wkb, Kp, Bc * Sc, Dc, Dc, Dc, Dc, Dc, 1.0f, nullptr);
    gemm_nt_wmma<__bf16, __bf16><<<gproj, blk, 0, stream>>>(
        vbf, wvb, Vp, Bc * Sc, Dc, Dc, Dc, Dc, Dc, 1.0f, nullptr);

    {
        const size_t total = (size_t)Bc * Hc * DKc * Sc;
        transpose_v_heads<<<dim3((unsigned)((total + 255) / 256)), blk, 0, stream>>>(
            Vp, Vt, Bc, Hc, Sc, Dc, DKc);
    }

    const dim3 gattn(1, Sc / 128, Bc * Hc);
    attn_fused<<<gattn, blk, 0, stream>>>(Qp, Kp, Vt, attn, ctx, mask, 0.125f);

    gemm_nt_wmma<__bf16, float><<<gproj, blk, 0, stream>>>(
        ctx, wob, out, Bc * Sc, Dc, Dc, Dc, Dc, Dc, 1.0f, bo);
}
